// Wav2Vec2GumbelVectorQuantizer_15006615732880
// MI455X (gfx1250) — compile-verified, matrix operand order checked by reading, not run on hardware
//
#include <hip/hip_runtime.h>

// ---------------------------------------------------------------------------
// Problem constants (from reference): B=8, T=4096, D=1024, G=1, V=1024
// ---------------------------------------------------------------------------
constexpr int Bc = 8;
constexpr int Tc = 4096;
constexpr int Dc = 1024;
constexpr int Vc = 1024;

constexpr long long Q_ELEMS = (long long)Bc * Tc * Dc;   // 33554432
constexpr long long PP_OFF  = Q_ELEMS;                   // scalar perplexity
constexpr long long IDX_OFF = Q_ELEMS + 1;               // [B,T] indices (as float)

typedef __attribute__((ext_vector_type(16))) __bf16 v16bf;
typedef __attribute__((ext_vector_type(8)))  float  v8f;

union BFrag { v16bf v; uint4 u[2]; };

// LDS layout (dynamic):
//  Lg     : 32*1024 f32          (131072 B)  logits tile
//  Abuf   : 32*32   bf16         (  2048 B)  A (z) staging tile
//  red    : 32*8    f32          (  1024 B)  partial reductions
//  redi   : 32*8    i32          (  1024 B)
//  rowmax : 32 f32, rowsum: 32 f32, gidx: 32 i32, tsum: 4 f32
constexpr unsigned OFF_ABUF   = 32 * 1024 * 4;
constexpr unsigned OFF_RED    = OFF_ABUF + 32 * 32 * 2;
constexpr unsigned OFF_REDI   = OFF_RED + 32 * 8 * 4;
constexpr unsigned OFF_ROWMAX = OFF_REDI + 32 * 8 * 4;
constexpr unsigned OFF_ROWSUM = OFF_ROWMAX + 32 * 4;
constexpr unsigned OFF_GIDX   = OFF_ROWSUM + 32 * 4;
constexpr unsigned OFF_TSUM   = OFF_GIDX + 32 * 4;
constexpr unsigned SMEM_BYTES = OFF_TSUM + 4 * 4;

// ---------------------------------------------------------------------------
// Kernel 0: convert proj_w (f32, [V,D] row-major) -> bf16 in workspace.
// 2 MB result stays resident in the 192 MB L2 for the whole GEMM.
// ---------------------------------------------------------------------------
__global__ void convert_w_bf16(const float* __restrict__ w,
                               __bf16* __restrict__ wbf) {
    int i = (blockIdx.x * blockDim.x + threadIdx.x) * 4;
    float4 f = *(const float4*)(w + i);
    alignas(8) __bf16 h[4] = { (__bf16)f.x, (__bf16)f.y, (__bf16)f.z, (__bf16)f.w };
    *(uint2*)(wbf + i) = *(const uint2*)h;
}

__global__ void zero_pp(float* out) { out[PP_OFF] = 0.0f; }

// ---------------------------------------------------------------------------
// Fused kernel: bf16 WMMA GEMM (32 rows x 1024 cols, K=1024) + softmax stats,
// argmax indices, gumbel argmax, perplexity partials, embedding gather.
// One WG = all 8 batches x 4 consecutive t  -> perplexity is intra-WG.
// ---------------------------------------------------------------------------
__global__ __launch_bounds__(256) void vq_fused(
    const float* __restrict__ z,       // [B,T,D]
    const float* __restrict__ gu,      // [B,T,V]
    const __bf16* __restrict__ wbf,    // [V,D] bf16
    const float* __restrict__ bias,    // [V]
    const float* __restrict__ emb,     // [V,D]
    float* __restrict__ out)
{
    extern __shared__ char smem[];
    float*  Lg     = (float*)smem;
    __bf16* Abuf   = (__bf16*)(smem + OFF_ABUF);
    float*  red    = (float*)(smem + OFF_RED);
    int*    redi   = (int*)(smem + OFF_REDI);
    float*  rowmax = (float*)(smem + OFF_ROWMAX);
    float*  rowsum = (float*)(smem + OFF_ROWSUM);
    int*    gidx   = (int*)(smem + OFF_GIDX);
    float*  tsum   = (float*)(smem + OFF_TSUM);

    const int tid  = threadIdx.x;
    const int t0   = blockIdx.x * 4;

    // ---- GEMM thread mapping -------------------------------------------
    const int lane  = tid & 31;
    const int wv    = tid >> 5;          // wave 0..7 -> column block of 128
    const int n0    = wv * 128;
    const int khalf = (lane & 16) ? 8 : 0;  // WMMA 16-bit K-half select
    const int mloc  = lane & 15;            // row (A) / column (B) within tile

    // ---- A staging mapping: row rl = tl*8 + b --------------------------
    const int arow = tid >> 3;           // 0..31
    const int kp   = (tid & 7) * 4;      // 4 floats per thread per K step
    const int ab   = arow & 7;
    const int atl  = arow >> 3;
    const float* zrow = z + (size_t)(ab * Tc + t0 + atl) * Dc;

    v8f acc[2][8] = {};

    for (int kb = 0; kb < Dc; kb += 32) {
        __syncthreads();
        {   // load 32x32 f32 of z, convert to bf16, stage in LDS
            float4 f = *(const float4*)(zrow + kb + kp);
            alignas(8) __bf16 h[4] = { (__bf16)f.x, (__bf16)f.y,
                                       (__bf16)f.z, (__bf16)f.w };
            *(uint2*)(Abuf + arow * 32 + kp) = *(const uint2*)h;
        }
        __syncthreads();

        BFrag a0, a1;
        {   // A fragments: lane<16 holds K 0..7 & 16..23; lane>=16 the other halves
            const uint4* pa0 = (const uint4*)(Abuf + (mloc) * 32 + khalf);
            a0.u[0] = pa0[0]; a0.u[1] = pa0[2];
            const uint4* pa1 = (const uint4*)(Abuf + (16 + mloc) * 32 + khalf);
            a1.u[0] = pa1[0]; a1.u[1] = pa1[2];
        }
#pragma unroll
        for (int nt = 0; nt < 8; ++nt) {
            // B fragment: lane = output column, rows of W are contiguous in K
            BFrag bf;
            const uint4* pw = (const uint4*)(wbf + (size_t)(n0 + nt * 16 + mloc) * Dc
                                             + kb + khalf);
            bf.u[0] = pw[0]; bf.u[1] = pw[2];
            acc[0][nt] = __builtin_amdgcn_wmma_f32_16x16x32_bf16(
                false, a0.v, false, bf.v, (short)0, acc[0][nt], false, false);
            acc[1][nt] = __builtin_amdgcn_wmma_f32_16x16x32_bf16(
                false, a1.v, false, bf.v, (short)0, acc[1][nt], false, false);
        }
    }

    // ---- write logits (+bias) to LDS -----------------------------------
    const int rhalf = (lane & 16) ? 8 : 0;   // C/D layout: lanes>=16 are rows+8
#pragma unroll
    for (int nt = 0; nt < 8; ++nt) {
        int col = n0 + nt * 16 + mloc;
        float bv = bias[col];
#pragma unroll
        for (int mt = 0; mt < 2; ++mt)
#pragma unroll
            for (int vi = 0; vi < 8; ++vi)
                Lg[(mt * 16 + rhalf + vi) * 1024 + col] = acc[mt][nt][vi] + bv;
    }
    if (tid < 4) tsum[tid] = 0.0f;
    __syncthreads();

    // ---- epilogue: 8 threads per row, 128 cols each ---------------------
    const int erow  = tid >> 3;
    const int epart = tid & 7;
    const int eb    = erow & 7;
    const int etl   = erow >> 3;
    const size_t grow = (size_t)(eb * Tc + t0 + etl);
    const float* Lr = Lg + erow * 1024;

    // Phase 1: max + argmax of logits -> indices output
    {
        float m = -__builtin_inff(); int mi = 0;
        for (int v = epart * 128; v < epart * 128 + 128; ++v) {
            float x = Lr[v];
            if (x > m) { m = x; mi = v; }
        }
        red[erow * 8 + epart] = m; redi[erow * 8 + epart] = mi;
    }
    __syncthreads();
    if (epart == 0) {
        float bm = red[erow * 8]; int bi = redi[erow * 8];
        for (int p = 1; p < 8; ++p) {
            float x = red[erow * 8 + p];
            if (x > bm) { bm = x; bi = redi[erow * 8 + p]; }
        }
        rowmax[erow] = bm;
        out[IDX_OFF + grow] = (float)bi;
    }
    __syncthreads();

    // Phase 2: sum exp(logit - max)
    {
        float rm = rowmax[erow], s = 0.0f;
        for (int v = epart * 128; v < epart * 128 + 128; ++v)
            s += __expf(Lr[v] - rm);
        red[erow * 8 + epart] = s;
    }
    __syncthreads();
    if (epart == 0) {
        float t = 0.0f;
        for (int p = 0; p < 8; ++p) t += red[erow * 8 + p];
        rowsum[erow] = t;
    }
    __syncthreads();

    // Phase 3: gumbel-perturbed argmax -> gather index
    {
        const float* gur = gu + grow * Vc;
        float gm = -__builtin_inff(); int gi = 0;
        for (int v = epart * 128; v < epart * 128 + 128; ++v) {
            float u = gur[v];
            u = fminf(fmaxf(u, 1e-6f), 1.0f - 1e-6f);
            float g = Lr[v] - __logf(-__logf(u));
            if (g > gm) { gm = g; gi = v; }
        }
        red[erow * 8 + epart] = gm; redi[erow * 8 + epart] = gi;
    }
    __syncthreads();
    if (epart == 0) {
        float bm = red[erow * 8]; int bi = redi[erow * 8];
        for (int p = 1; p < 8; ++p) {
            float x = red[erow * 8 + p];
            if (x > bm) { bm = x; bi = redi[erow * 8 + p]; }
        }
        gidx[erow] = bi;
    }
    __syncthreads();

    // Phase 4: perplexity: dist[t,v] = mean_b probs; accumulate sum_v dist^2
    {
        const int tl = tid >> 6;            // 0..3 (t within WG)
        const int vp = tid & 63;            // 64 column chunks of 16
        float rms[8], rss[8];
#pragma unroll
        for (int b = 0; b < 8; ++b) {
            rms[b] = rowmax[tl * 8 + b];
            rss[b] = 1.0f / rowsum[tl * 8 + b];
        }
        float part2 = 0.0f;
        for (int v = vp * 16; v < vp * 16 + 16; ++v) {
            float d = 0.0f;
#pragma unroll
            for (int b = 0; b < 8; ++b)
                d += __expf(Lg[(tl * 8 + b) * 1024 + v] - rms[b]) * rss[b];
            d *= 0.125f;
            part2 += d * d;
        }
        atomicAdd(&tsum[tl], part2);
    }
    __syncthreads();
    if (tid < 4) {
        float S = tsum[tid];
        atomicAdd(&out[PP_OFF], (1.0f / (S + 1e-10f)) * (1.0f / (float)Tc));
    }

    // Phase 5: quantized = embedding[gumbel-argmax] (forward value of y_hard)
    {
        int j = gidx[erow];
        const float4* src = (const float4*)(emb + (size_t)j * Dc) + epart * 32;
        float4*       dst = (float4*)(out + grow * Dc) + epart * 32;
#pragma unroll 8
        for (int i = 0; i < 32; ++i) dst[i] = src[i];
    }
}

// ---------------------------------------------------------------------------
extern "C" void kernel_launch(void* const* d_in, const int* in_sizes, int n_in,
                              void* d_out, int out_size, void* d_ws, size_t ws_size,
                              hipStream_t stream) {
    const float* z   = (const float*)d_in[0];
    const float* gu  = (const float*)d_in[1];
    const float* pw  = (const float*)d_in[2];
    const float* pb  = (const float*)d_in[3];
    const float* emb = (const float*)d_in[4];
    float* out = (float*)d_out;
    __bf16* wbf = (__bf16*)d_ws;   // 1024*1024 bf16 = 2 MB scratch

    // W -> bf16 (1M elements, 4 per thread)
    convert_w_bf16<<<1024, 256, 0, stream>>>(pw, wbf);
    zero_pp<<<1, 1, 0, stream>>>(out);
    // 4096/4 = 1024 workgroups, each: 8 batches x 4 t rows, full V
    vq_fused<<<Tc / 4, 256, SMEM_BYTES, stream>>>(z, gu, wbf, pb, emb, out);
}
